// ConfidenceCalibration_44985487458449
// MI455X (gfx1250) — matrix operand order repaired and checked by reading, not compiled, not run on hardware
//
#include <hip/hip_runtime.h>
#include <math.h>

typedef __attribute__((ext_vector_type(16))) _Float16 v16h;
typedef __attribute__((ext_vector_type(8)))  _Float16 v8h;
typedef __attribute__((ext_vector_type(4)))  _Float16 v4h;
typedef __attribute__((ext_vector_type(8)))  float    v8f;
typedef __attribute__((ext_vector_type(4)))  float    v4f;

#define D_DIM   1024
#define H_DIM   512
#define NBINS   15
#define M_TILE  16
#define HPAD    (H_DIM + 4)     // row stride 2064 B: 16-B aligned, bank-skewed
#define LN_EPS  1e-5f

// ---------------------------------------------------------------------------
// Pre-pass: convert w1 [H,D] f32 -> f16 into workspace (1 MB). L2-resident
// afterwards; every GEMM block streams B-fragments from it.
// ---------------------------------------------------------------------------
__global__ void cvt_w1_f16_kernel(const float4* __restrict__ src,
                                  _Float16* __restrict__ dst, int n4) {
    int i = blockIdx.x * blockDim.x + threadIdx.x;
    if (i < n4) {
        float4 v = src[i];
        union { v4h v; unsigned long long u; } o;
        o.v[0] = (_Float16)v.x; o.v[1] = (_Float16)v.y;
        o.v[2] = (_Float16)v.z; o.v[3] = (_Float16)v.w;
        *(unsigned long long*)(dst + (size_t)i * 4) = o.u;
    }
}

// ---------------------------------------------------------------------------
// Fused: GEMM1 (WMMA f16->f32) + bias + LayerNorm + GELU(erf) + GEMM2 +
// softmax + sigmoid(mean(x)) + bucketize + weighted sum + clip.
// One block = 16 rows of x. 8 waves, each owns 64 of the 512 h-columns.
// ---------------------------------------------------------------------------
__global__ __launch_bounds__(256)
void ConfidenceCalibration_44985487458449_kernel(
        const float*    __restrict__ x,      // [B, 1024]
        const _Float16* __restrict__ w1h,    // [512, 1024] f16 (workspace)
        const float*    __restrict__ b1,     // [512]
        const float*    __restrict__ ln_g,   // [512]
        const float*    __restrict__ ln_b,   // [512]
        const float*    __restrict__ w2,     // [15, 512]
        const float*    __restrict__ b2,     // [15]
        const float*    __restrict__ bin_scaling, // [15]
        float*          __restrict__ out)    // [B]
{
    __shared__ float h_lds[M_TILE][HPAD];      // h tile (f32), ~33 KB
    __shared__ float s_rowsum[M_TILE];         // sum of x per row (for base)
    __shared__ float s_mu[M_TILE], s_rs[M_TILE];
    __shared__ float s_logits[M_TILE][NBINS];

    const int tid  = threadIdx.x;
    const int wave = tid >> 5;        // 0..7
    const int lane = tid & 31;
    const int mfr  = lane & 15;       // fragment row (A) / col (B,C)
    const int khi  = lane >> 4;       // 0/1 half-lane group
    const int rowBase = blockIdx.x * M_TILE;
    const int colBase = wave * 64;

    // ---------------- GEMM1: h = x @ w1^T ----------------
    v8f c[4] = {v8f{}, v8f{}, v8f{}, v8f{}};
    const float* xrow = x + (size_t)(rowBase + mfr) * D_DIM;
    v4f xs = {0.f, 0.f, 0.f, 0.f};   // 4 independent partial-sum chains

    for (int k = 0; k < D_DIM; k += 32) {
        __builtin_prefetch((const void*)(xrow + k + 64), 0, 0);

        // A fragment (16x32 f16): lanes 0-15 hold K {k..k+7, k+16..k+23},
        // lanes 16-31 hold K {k+8..k+15, k+24..k+31}.
        const int ka = k + khi * 8;
        v4f a0 = *(const v4f*)(xrow + ka);
        v4f a1 = *(const v4f*)(xrow + ka + 4);
        v4f a2 = *(const v4f*)(xrow + ka + 16);
        v4f a3 = *(const v4f*)(xrow + ka + 20);
        xs += a0; xs += a1; xs += a2; xs += a3;

        union { v16h v; _Float16 e[16]; } a;
        #pragma unroll
        for (int j = 0; j < 4; ++j) {
            a.e[j]      = (_Float16)a0[j];
            a.e[j + 4]  = (_Float16)a1[j];
            a.e[j + 8]  = (_Float16)a2[j];
            a.e[j + 12] = (_Float16)a3[j];
        }

        // B fragments (32x16 f16): lane n<16 holds K {k..k+7, k+16..k+23} of column n,
        // lanes 16-31 hold K {k+8..k+15, k+24..k+31} (matches A).
        #pragma unroll
        for (int ct = 0; ct < 4; ++ct) {
            const _Float16* wrow =
                w1h + (size_t)(colBase + ct * 16 + mfr) * D_DIM + k + khi * 8;
            union { v16h v; v8h h[2]; } b;
            b.h[0] = *(const v8h*)(wrow);
            b.h[1] = *(const v8h*)(wrow + 16);
            c[ct] = __builtin_amdgcn_wmma_f32_16x16x32_f16(
                        false, a.v, false, b.v, (short)0, c[ct], false, false);
        }
    }

    // Row sums of x fall out of the A loads: lane and lane^16 together cover
    // the full 1024-wide row.  All 8 waves loaded identical A; wave 0 writes.
    float xsum = (xs[0] + xs[1]) + (xs[2] + xs[3]);
    xsum += __shfl_xor(xsum, 16);
    if (wave == 0 && lane < 16) s_rowsum[lane] = xsum;

    // Spill h (+ b1) to LDS.  C layout: lane<16 -> M=r, lane>=16 -> M=r+8.
    #pragma unroll
    for (int ct = 0; ct < 4; ++ct) {
        const int col  = colBase + ct * 16 + mfr;
        const float bias = b1[col];
        #pragma unroll
        for (int r = 0; r < 8; ++r)
            h_lds[r + khi * 8][col] = c[ct][r] + bias;
    }
    __syncthreads();

    // ---------------- LayerNorm stats (16 threads per row) ----------------
    {
        const int row = tid >> 4, seg = tid & 15;
        v4f s4 = {0.f,0.f,0.f,0.f}, q4 = {0.f,0.f,0.f,0.f};
        #pragma unroll
        for (int j = 0; j < 8; ++j) {
            v4f v = *(const v4f*)&h_lds[row][seg * 32 + j * 4];
            s4 += v; q4 += v * v;
        }
        float s  = (s4[0] + s4[1]) + (s4[2] + s4[3]);
        float s2 = (q4[0] + q4[1]) + (q4[2] + q4[3]);
        #pragma unroll
        for (int m = 8; m >= 1; m >>= 1) {      // stays within 16-lane half
            s  += __shfl_xor(s,  m);
            s2 += __shfl_xor(s2, m);
        }
        if (seg == 0) {
            float mu  = s  * (1.0f / H_DIM);
            float var = s2 * (1.0f / H_DIM) - mu * mu;
            s_mu[row] = mu;
            s_rs[row] = rsqrtf(var + LN_EPS);
        }
    }
    __syncthreads();

    // ---------------- Normalize + exact GELU (vectorized) ----------------
    {
        const int row = tid >> 4, seg = tid & 15;
        const float mu = s_mu[row], rs = s_rs[row];
        #pragma unroll
        for (int j = 0; j < 8; ++j) {
            const int cidx = seg * 32 + j * 4;
            v4f v = *(const v4f*)&h_lds[row][cidx];
            v4f g = *(const v4f*)(ln_g + cidx);
            v4f bb = *(const v4f*)(ln_b + cidx);
            #pragma unroll
            for (int e = 0; e < 4; ++e) {
                float t = (v[e] - mu) * rs * g[e] + bb[e];
                v[e] = 0.5f * t * (1.0f + erff(t * 0.70710678118654752f));
            }
            *(v4f*)&h_lds[row][cidx] = v;
        }
    }
    __syncthreads();

    // ---------------- GEMM2: logits[16,15] (b128 LDS reads) ----------------
    if (tid < M_TILE * NBINS) {
        const int row = tid / NBINS, bin = tid % NBINS;
        const float* w2r = w2 + (size_t)bin * H_DIM;
        v4f acc4 = {0.f,0.f,0.f,0.f};
        #pragma unroll 4
        for (int j = 0; j < H_DIM; j += 4) {
            v4f hv = *(const v4f*)&h_lds[row][j];
            v4f wv = *(const v4f*)(w2r + j);
            acc4 += hv * wv;
        }
        s_logits[row][bin] = b2[bin] + (acc4[0] + acc4[1]) + (acc4[2] + acc4[3]);
    }
    __syncthreads();

    // ---------------- softmax + base/bucketize + output ----------------
    if (tid < M_TILE) {
        const int row = tid;
        // base = sigmoid(mean(x_row))
        const float base = 1.0f / (1.0f + expf(-(s_rowsum[row] * (1.0f / D_DIM))));
        // bucketize on uniform [0,1] grid of 16 boundaries (searchsorted-right)
        int idx = (int)floorf(base * (float)NBINS);
        const float scale = (idx >= 0 && idx < NBINS) ? bin_scaling[idx] : 0.0f;
        const float calib = base * scale;

        float mx = s_logits[row][0];
        #pragma unroll
        for (int n = 1; n < NBINS; ++n) mx = fmaxf(mx, s_logits[row][n]);
        float den = 0.0f, e[NBINS];
        #pragma unroll
        for (int n = 0; n < NBINS; ++n) { e[n] = expf(s_logits[row][n] - mx); den += e[n]; }
        const float inv = 1.0f / den;
        float fin = 0.0f;
        #pragma unroll
        for (int n = 0; n < NBINS; ++n) fin = fmaf(calib, e[n] * inv, fin);

        out[rowBase + row] = fminf(fmaxf(fin, 0.0f), 1.0f);
    }
}

// ---------------------------------------------------------------------------
extern "C" void kernel_launch(void* const* d_in, const int* in_sizes, int n_in,
                              void* d_out, int out_size, void* d_ws, size_t ws_size,
                              hipStream_t stream) {
    const float* x     = (const float*)d_in[0];
    const float* w1    = (const float*)d_in[1];
    const float* b1    = (const float*)d_in[2];
    const float* ln_g  = (const float*)d_in[3];
    const float* ln_b  = (const float*)d_in[4];
    const float* w2    = (const float*)d_in[5];
    const float* b2    = (const float*)d_in[6];
    const float* bscal = (const float*)d_in[7];
    float* out = (float*)d_out;

    _Float16* w1h = (_Float16*)d_ws;                  // 512*1024*2 = 1 MB

    const int B = in_sizes[0] / D_DIM;                // 32768

    {   // w1 f32 -> f16
        const int n4 = (H_DIM * D_DIM) / 4;
        cvt_w1_f16_kernel<<<(n4 + 255) / 256, 256, 0, stream>>>(
            (const float4*)w1, w1h, n4);
    }

    ConfidenceCalibration_44985487458449_kernel<<<B / M_TILE, 256, 0, stream>>>(
        x, w1h, b1, ln_g, ln_b, w2, b2, bscal, out);
}
